// MultiHeadAttention_69664369541436
// MI455X (gfx1250) — matrix operand order repaired and checked by reading, not compiled, not run on hardware
//
#include <hip/hip_runtime.h>
#include <hip/hip_bf16.h>

typedef __attribute__((ext_vector_type(16))) __bf16 v16bf;
typedef __attribute__((ext_vector_type(8)))  __bf16 v8bf;
typedef __attribute__((ext_vector_type(8)))  float  v8f;
typedef int v4i __attribute__((vector_size(16)));   // pointee type the async builtin expects

#define T_    1024
#define SCALE 0.125f   // 1/sqrt(64)

// ---- CDNA5 async global->LDS path (probe-gated; falls back to sync copies) ----
#if __has_builtin(__builtin_amdgcn_global_load_async_to_lds_b128) && \
    __has_builtin(__builtin_amdgcn_s_wait_asynccnt)
#define HAVE_ASYNC 1
#define WAIT_ASYNC(n) __builtin_amdgcn_s_wait_asynccnt(n)
#else
#define HAVE_ASYNC 0
#define WAIT_ASYNC(n) ((void)0)
#endif

__device__ inline void cp16(__bf16* lds, const __bf16* g) {
#if HAVE_ASYNC
  // generic LDS pointer low 32 bits == LDS byte address (ISA aperture rule)
  __builtin_amdgcn_global_load_async_to_lds_b128(
      (__attribute__((address_space(1))) v4i*)(unsigned long long)(const void*)g,
      (__attribute__((address_space(3))) v4i*)(unsigned)(unsigned long long)(void*)lds,
      0, 0);
#else
  *(v8bf*)lds = *(const v8bf*)g;
#endif
}

__device__ inline v8f wmma_bf16(v16bf a, v16bf b, v8f c) {
  return __builtin_amdgcn_wmma_f32_16x16x32_bf16(false, a, false, b, (short)0, c, false, false);
}
__device__ inline v16bf mk16(v8bf lo, v8bf hi) {
  v16bf r;
#pragma unroll
  for (int j = 0; j < 8; ++j) { r[j] = lo[j]; r[j + 8] = hi[j]; }
  return r;
}
__device__ inline v8f zero8() {
  v8f z;
#pragma unroll
  for (int i = 0; i < 8; ++i) z[i] = 0.0f;
  return z;
}

// ---------------- f32 -> bf16 convert (with zero tail padding) ----------------
__global__ __launch_bounds__(256) void cvt_kernel(const float* __restrict__ in,
                                                  __bf16* __restrict__ out,
                                                  int n, int ntot) {
  int i = blockIdx.x * 256 + threadIdx.x;
  if (i < ntot) out[i] = (__bf16)((i < n) ? in[i] : 0.0f);
}

// ---------------- f32 [K,N] -> bf16 transposed [N,K] ----------------
__global__ __launch_bounds__(256) void cvt_transpose_kernel(const float* __restrict__ in,
                                                            __bf16* __restrict__ out,
                                                            int K, int N) {
  __shared__ float tile[32][33];
  int n0 = blockIdx.x * 32, k0 = blockIdx.y * 32;
  int tx = threadIdx.x & 31, ty = threadIdx.x >> 5;
#pragma unroll
  for (int r = 0; r < 32; r += 8)
    tile[ty + r][tx] = in[(size_t)(k0 + ty + r) * N + n0 + tx];
  __syncthreads();
#pragma unroll
  for (int r = 0; r < 32; r += 8)
    out[(size_t)(n0 + ty + r) * K + k0 + tx] = (__bf16)tile[tx][ty + r];
}

// ---------------- bf16 WMMA GEMM: C = A(MxK) * B(KxN), B given transposed [N,K] ----------------
// mode 0: store bf16 head-major [B,H,T,DH] (Q/K/V projections)
// mode 1: store f32 row-major with +bias[col] +resid[row,col] (output proj)
#define BM 128
#define BN 64
#define BK 64
#define AP 72
#define BP 72

__global__ __launch_bounds__(256) void gemm_bf16_kernel(
    const __bf16* __restrict__ A, const __bf16* __restrict__ BT,
    void* __restrict__ Cout, int M, int N, int K, int mode,
    const float* __restrict__ bias, const float* __restrict__ resid) {
  __shared__ __bf16 As[2][BM * AP];   // 2 x 18432 B
  __shared__ __bf16 Bs[2][BN * BP];   // 2 x  9216 B  (Bs[n][k])
  const int tid  = threadIdx.x;
  const int lane = tid & 31;
  const int wid  = tid >> 5;
  const int wm = wid & 3, wn = wid >> 2;
  const int lm = lane & 15, lh = lane >> 4;
  const int n0 = blockIdx.x * BN;
  const int m0 = blockIdx.y * BM;

  v8f acc[2][2];
#pragma unroll
  for (int a = 0; a < 2; ++a)
#pragma unroll
    for (int b = 0; b < 2; ++b) acc[a][b] = zero8();

  // stage one BKx(BM|BN) tile pair into LDS buffer `buf` (async when available)
  auto stage = [&](int kt, int buf) {
    int k0 = kt * BK;
#pragma unroll
    for (int p = 0; p < 4; ++p) {            // A: 128 rows x 64 cols = 1024 16B chunks
      int idx = p * 256 + tid;
      int row = idx >> 3;
      int col = (idx & 7) * 8;
      cp16(&As[buf][row * AP + col], A + (size_t)(m0 + row) * K + k0 + col);
    }
#pragma unroll
    for (int p = 0; p < 2; ++p) {            // B^T: 64 rows x 64 cols = 512 chunks
      int idx = p * 256 + tid;
      int row = idx >> 3;
      int col = (idx & 7) * 8;
      cp16(&Bs[buf][row * BP + col], BT + (size_t)(n0 + row) * K + k0 + col);
    }
  };

  const int NT = K / BK;
  stage(0, 0);
  for (int it = 0; it < NT; ++it) {
    int cur = it & 1;
    if (it + 1 < NT) { stage(it + 1, cur ^ 1); WAIT_ASYNC(6); }
    else             { WAIT_ASYNC(0); }
    __syncthreads();
#pragma unroll
    for (int s = 0; s < 2; ++s) {            // two K=32 steps per tile
      v16bf af[2], bfg[2];
#pragma unroll
      for (int mt = 0; mt < 2; ++mt) {
        const __bf16* ap = &As[cur][(wm * 32 + mt * 16 + lm) * AP + s * 32 + lh * 16];
        af[mt] = mk16(*(const v8bf*)ap, *(const v8bf*)(ap + 8));
      }
#pragma unroll
      for (int nt = 0; nt < 2; ++nt) {
        const __bf16* bp = &Bs[cur][(wn * 32 + nt * 16 + lm) * BP + s * 32 + lh * 16];
        bfg[nt] = mk16(*(const v8bf*)bp, *(const v8bf*)(bp + 8));
      }
#pragma unroll
      for (int mt = 0; mt < 2; ++mt)
#pragma unroll
        for (int nt = 0; nt < 2; ++nt)
          acc[mt][nt] = wmma_bf16(af[mt], bfg[nt], acc[mt][nt]);
    }
    __syncthreads();
  }

#pragma unroll
  for (int mt = 0; mt < 2; ++mt) {
#pragma unroll
    for (int nt = 0; nt < 2; ++nt) {
#pragma unroll
      for (int i = 0; i < 8; ++i) {
        int m = m0 + wm * 32 + mt * 16 + i + 8 * lh;
        int c = n0 + wn * 32 + nt * 16 + lm;
        float v = acc[mt][nt][i];
        if (mode == 0) {
          int b = m >> 10, t = m & (T_ - 1);
          int h = c >> 6, d = c & 63;
          ((__bf16*)Cout)[(((size_t)(b * 16 + h)) * T_ + t) * 64 + d] = (__bf16)v;
        } else {
          ((float*)Cout)[(size_t)m * N + c] =
              v + bias[c] + resid[(size_t)m * N + c];
        }
      }
    }
  }
}

// ---------------- flash attention with relative-position skew ----------------
// one wave32 per (b,h, 16 query rows). skew(QE)[q,k] = QE[q, k-q+T-1]
#define QEP  1072   // band pitch (bf16 elems); band width used = 1056
#define NTQE 66
__global__ __launch_bounds__(32) void attn_kernel(
    const __bf16* __restrict__ Qg, const __bf16* __restrict__ Kg,
    const __bf16* __restrict__ Vg, const __bf16* __restrict__ Eb,
    const unsigned char* __restrict__ mask, __bf16* __restrict__ Og) {
  __shared__ __bf16 qe[16 * QEP];  // 34304 B : QE band, rows m=0..15
  __shared__ __bf16 Pl[16 * 40];   //  1280 B : P tile route to A-frag layout
  __shared__ __bf16 Vt[64 * 40];   //  5120 B : V tile transposed [d][kk]

  const int lane = threadIdx.x;
  const int lm = lane & 15, lh = lane >> 4;
  const int bh = blockIdx.x;
  const int b = bh >> 4, h = bh & 15;
  const int q0 = blockIdx.y * 16;

  // Q A-fragments, two K=32 halves of d=64
  v16bf qa[2];
  {
    const __bf16* qp = Qg + ((size_t)bh * T_ + (q0 + lm)) * 64 + lh * 16;
#pragma unroll
    for (int s = 0; s < 2; ++s)
      qa[s] = mk16(*(const v8bf*)(qp + s * 32), *(const v8bf*)(qp + s * 32 + 8));
  }

  // QE band precompute: r in [r_lo, r_lo+1056), r = k - q + (T-1)
  const int r_lo = (T_ - 1) - (q0 + 15);
#pragma unroll 1
  for (int nt = 0; nt < NTQE; ++nt) {
    int r = r_lo + nt * 16 + lm;                 // E padded to 2112 rows
    const __bf16* ep = Eb + (size_t)r * 64 + lh * 16;
    v8f a = zero8();
#pragma unroll
    for (int s = 0; s < 2; ++s) {
      v16bf eb = mk16(*(const v8bf*)(ep + s * 32), *(const v8bf*)(ep + s * 32 + 8));
      a = wmma_bf16(qa[s], eb, a);
    }
#pragma unroll
    for (int i = 0; i < 8; ++i) {
      int m = i + 8 * lh;
      qe[m * QEP + nt * 16 + lm] = (__bf16)a[i];
    }
  }
  __syncthreads();

  float rm[8], rs[8];
  v8f o[4];
#pragma unroll
  for (int i = 0; i < 8; ++i) { rm[i] = -1e30f; rs[i] = 0.0f; }
#pragma unroll
  for (int dt = 0; dt < 4; ++dt) o[dt] = zero8();

  const size_t mbase = (size_t)b * T_ * T_;

#pragma unroll 1
  for (int kt = 0; kt < T_ / 32; ++kt) {
    int k0 = kt * 32;
    // stage V tile transposed: Vt[d][kk_local]
    {
      const __bf16* vrow = Vg + ((size_t)bh * T_ + (k0 + lane)) * 64;
#pragma unroll
      for (int c = 0; c < 8; ++c) {
        v8bf t = *(const v8bf*)(vrow + c * 8);
#pragma unroll
        for (int j = 0; j < 8; ++j) Vt[(c * 8 + j) * 40 + lane] = t[j];
      }
    }
    // S = Q * K^T  (B-fragments straight from global: K rows are contiguous)
    v8f s0 = zero8(), s1 = zero8();
#pragma unroll
    for (int s = 0; s < 2; ++s) {
      const __bf16* kp0 = Kg + ((size_t)bh * T_ + (k0 + lm)) * 64 + s * 32 + lh * 16;
      const __bf16* kp1 = kp0 + 16 * 64;
      v16bf b0 = mk16(*(const v8bf*)kp0, *(const v8bf*)(kp0 + 8));
      v16bf b1 = mk16(*(const v8bf*)kp1, *(const v8bf*)(kp1 + 8));
      s0 = wmma_bf16(qa[s], b0, s0);
      s1 = wmma_bf16(qa[s], b1, s1);
    }
    // + skewed rel, * scale, mask
    float x0[8], x1[8];
#pragma unroll
    for (int i = 0; i < 8; ++i) {
      int m = i + 8 * lh;
      int q_abs = q0 + m;
      int ka0 = k0 + lm;
      int ka1 = ka0 + 16;
      float r0 = (float)qe[m * QEP + (ka0 - m + 15)];   // band idx = k - m + 15
      float r1 = (float)qe[m * QEP + (ka1 - m + 15)];
      float v0 = (s0[i] + r0) * SCALE;
      float v1 = (s1[i] + r1) * SCALE;
      if (mask[mbase + (size_t)q_abs * T_ + ka0]) v0 = -1e9f;
      if (mask[mbase + (size_t)q_abs * T_ + ka1]) v1 = -1e9f;
      x0[i] = v0; x1[i] = v1;
    }
    // online softmax; row m lives in lanes of one 16-lane half
#pragma unroll
    for (int i = 0; i < 8; ++i) {
      float mx = fmaxf(x0[i], x1[i]);
#pragma unroll
      for (int off = 1; off < 16; off <<= 1) mx = fmaxf(mx, __shfl_xor(mx, off, 32));
      float mnew  = fmaxf(rm[i], mx);
      float alpha = __expf(rm[i] - mnew);
      float p0 = __expf(x0[i] - mnew);
      float p1 = __expf(x1[i] - mnew);
      float sum = p0 + p1;
#pragma unroll
      for (int off = 1; off < 16; off <<= 1) sum += __shfl_xor(sum, off, 32);
      rs[i] = rs[i] * alpha + sum;
      rm[i] = mnew;
#pragma unroll
      for (int dt = 0; dt < 4; ++dt) o[dt][i] *= alpha;
      int m = i + 8 * lh;
      Pl[m * 40 + lm]      = (__bf16)p0;
      Pl[m * 40 + 16 + lm] = (__bf16)p1;
    }
    __syncthreads();
    // O += P * V
    {
      const __bf16* pp = &Pl[lm * 40 + lh * 16];
      v16bf pa = mk16(*(const v8bf*)pp, *(const v8bf*)(pp + 8));
#pragma unroll
      for (int dt = 0; dt < 4; ++dt) {
        const __bf16* vp = &Vt[(dt * 16 + lm) * 40 + lh * 16];
        v16bf vb = mk16(*(const v8bf*)vp, *(const v8bf*)(vp + 8));
        o[dt] = wmma_bf16(pa, vb, o[dt]);
      }
    }
    __syncthreads();
  }

  // write attention output [B, T, H*DH] bf16
#pragma unroll
  for (int dt = 0; dt < 4; ++dt) {
#pragma unroll
    for (int i = 0; i < 8; ++i) {
      int m = i + 8 * lh;
      float val = o[dt][i] / rs[i];
      Og[((size_t)(b * T_ + q0 + m)) * 1024 + h * 64 + dt * 16 + lm] = (__bf16)val;
    }
  }
}

// ---------------- layernorm: out = (y-mu)/sqrt(var+eps)*gamma+beta ----------------
__global__ __launch_bounds__(256) void ln_kernel(const float* __restrict__ y,
                                                 const float* __restrict__ gamma,
                                                 const float* __restrict__ beta,
                                                 float* __restrict__ out) {
  __shared__ float s1[8], s2[8];
  int row = blockIdx.x;
  int tid = threadIdx.x;
  const float* yr = y + (size_t)row * 1024;
  float a = 0.f, b = 0.f;
  for (int c = tid; c < 1024; c += 256) { float v = yr[c]; a += v; b += v * v; }
#pragma unroll
  for (int off = 1; off < 32; off <<= 1) {
    a += __shfl_xor(a, off, 32);
    b += __shfl_xor(b, off, 32);
  }
  if ((tid & 31) == 0) { s1[tid >> 5] = a; s2[tid >> 5] = b; }
  __syncthreads();
  float ta = 0.f, tb = 0.f;
#pragma unroll
  for (int w = 0; w < 8; ++w) { ta += s1[w]; tb += s2[w]; }
  float mu   = ta * (1.0f / 1024.0f);
  float var  = tb * (1.0f / 1024.0f) - mu * mu;
  float rstd = rsqrtf(var + 1e-5f);
  for (int c = tid; c < 1024; c += 256)
    out[(size_t)row * 1024 + c] = (yr[c] - mu) * rstd * gamma[c] + beta[c];
}

// ---------------- host ----------------
extern "C" void kernel_launch(void* const* d_in, const int* in_sizes, int n_in,
                              void* d_out, int out_size, void* d_ws, size_t ws_size,
                              hipStream_t stream) {
  const float* x     = (const float*)d_in[0];
  const float* ctx   = (const float*)d_in[1];
  const unsigned char* mask = (const unsigned char*)d_in[2];
  const float* lut   = (const float*)d_in[3];
  const float* Wq    = (const float*)d_in[4];
  const float* Wk    = (const float*)d_in[5];
  const float* Wv    = (const float*)d_in[6];
  const float* Wo    = (const float*)d_in[7];
  const float* bo    = (const float*)d_in[8];
  const float* gamma = (const float*)d_in[9];
  const float* beta  = (const float*)d_in[10];

  char* ws = (char*)d_ws;
  size_t off = 0;
  auto take = [&](size_t bytes) -> char* {
    char* p = ws + off;
    off += (bytes + 255) & ~(size_t)255;
    return p;
  };
  __bf16* xb   = (__bf16*)take((size_t)4096 * 1024 * 2);   // x bf16
  __bf16* cb   = (__bf16*)take((size_t)4096 * 1024 * 2);   // context bf16
  __bf16* wqt  = (__bf16*)take((size_t)1024 * 1024 * 2);   // Wq^T [N,K]
  __bf16* wkt  = (__bf16*)take((size_t)1024 * 1024 * 2);
  __bf16* wvt  = (__bf16*)take((size_t)1024 * 1024 * 2);
  __bf16* wot  = (__bf16*)take((size_t)1024 * 1024 * 2);
  __bf16* eb   = (__bf16*)take((size_t)2112 * 64 * 2);     // E padded to 2112 rows
  __bf16* Qg   = (__bf16*)take((size_t)4096 * 1024 * 2);   // [B,H,T,DH]
  __bf16* Kg   = (__bf16*)take((size_t)4096 * 1024 * 2);
  __bf16* Vg   = (__bf16*)take((size_t)4096 * 1024 * 2);
  __bf16* aout = (__bf16*)take((size_t)4096 * 1024 * 2);   // [B,T,H*DH]
  float*  ypr  = (float*)take((size_t)4096 * 1024 * 4);    // proj + bias + resid

  // converts (+ one-time weight transposes for async-friendly GEMM B tiles)
  cvt_kernel<<<(4194304 + 255) / 256, 256, 0, stream>>>(x, xb, 4194304, 4194304);
  cvt_kernel<<<(4194304 + 255) / 256, 256, 0, stream>>>(ctx, cb, 4194304, 4194304);
  dim3 tg(32, 32);
  cvt_transpose_kernel<<<tg, 256, 0, stream>>>(Wq, wqt, 1024, 1024);
  cvt_transpose_kernel<<<tg, 256, 0, stream>>>(Wk, wkt, 1024, 1024);
  cvt_transpose_kernel<<<tg, 256, 0, stream>>>(Wv, wvt, 1024, 1024);
  cvt_transpose_kernel<<<tg, 256, 0, stream>>>(Wo, wot, 1024, 1024);
  cvt_kernel<<<(135168 + 255) / 256, 256, 0, stream>>>(lut, eb, 2047 * 64, 2112 * 64);

  // Q/K/V projections (head-major bf16 out)
  dim3 gg(1024 / BN, 4096 / BM);
  gemm_bf16_kernel<<<gg, 256, 0, stream>>>(xb, wqt, Qg, 4096, 1024, 1024, 0, nullptr, nullptr);
  gemm_bf16_kernel<<<gg, 256, 0, stream>>>(cb, wkt, Kg, 4096, 1024, 1024, 0, nullptr, nullptr);
  gemm_bf16_kernel<<<gg, 256, 0, stream>>>(cb, wvt, Vg, 4096, 1024, 1024, 0, nullptr, nullptr);

  // flash attention with skewed relative positions
  attn_kernel<<<dim3(64, 64), 32, 0, stream>>>(Qg, Kg, Vg, eb, mask, aout);

  // output projection + bias + residual (f32)
  gemm_bf16_kernel<<<gg, 256, 0, stream>>>(aout, wot, ypr, 4096, 1024, 1024, 1, bo, x);

  // layernorm -> final output
  ln_kernel<<<4096, 256, 0, stream>>>(ypr, gamma, beta, (float*)d_out);
}
